// AttentionBlock_39135742001195
// MI455X (gfx1250) — matrix operand order repaired and checked by reading, not compiled, not run on hardware
//
#include <hip/hip_runtime.h>

typedef __attribute__((ext_vector_type(16))) __bf16 v16bf;
typedef __attribute__((ext_vector_type(8)))  float  v8f;

#define WMMA_BF16(A_, B_, C_) \
  __builtin_amdgcn_wmma_f32_16x16x32_bf16(false, (A_), false, (B_), (short)0, (C_), false, false)

static constexpr int BSZ = 32;    // batch
static constexpr int CCH = 256;   // channels
static constexpr int NHW = 1024;  // H*W
static constexpr int O3  = 768;   // 3*C

__device__ __forceinline__ v8f vzero8() {
  v8f z;
#pragma unroll
  for (int i = 0; i < 8; ++i) z[i] = 0.f;
  return z;
}

// A fragment: 16x32 (M x K), M-major storage with leading dim ld.
// lane<16: elems 0..7 -> K=k0+0..7, elems 8..15 -> K=k0+16..23
// lane>=16: elems 0..7 -> K=k0+8..15, elems 8..15 -> K=k0+24..31
__device__ __forceinline__ v16bf frag_a(const __bf16* base, int row0, int k0, int ld) {
  int lane = threadIdx.x & 31;
  const __bf16* p = base + (size_t)(row0 + (lane & 15)) * ld + k0 + ((lane & 16) ? 8 : 0);
  union { uint4 q[2]; v16bf v; } u;
  u.q[0] = *(const uint4*)(p);
  u.q[1] = *(const uint4*)(p + 16);
  return u.v;
}

// B fragment: 32x16 (K x N) loaded from BT stored N-major [N][K] with leading dim ld.
// lane<16: elems 0..7 -> K=k0+0..7, 8..15 -> K=k0+16..23 ; lane>=16: K=k0+8..15, k0+24..31 ; N = col0 + (lane&15)
__device__ __forceinline__ v16bf frag_b(const __bf16* base, int col0, int k0, int ld) {
  int lane = threadIdx.x & 31;
  const __bf16* p = base + (size_t)(col0 + (lane & 15)) * ld + k0 + ((lane & 16) ? 8 : 0);
  union { uint4 q[2]; v16bf v; } u;
  u.q[0] = *(const uint4*)(p);
  u.q[1] = *(const uint4*)(p + 16);
  return u.v;
}

// ---------------- small helper kernels ----------------

__global__ void cvt_bf16_kernel(const float* __restrict__ src, __bf16* __restrict__ dst, int n) {
  int i = blockIdx.x * 256 + threadIdx.x;
  if (i < n) dst[i] = (__bf16)src[i];
}

// GroupNorm statistics + SE pooled means. grid = B*32 groups, block = 256 (8 waves = 8 channels)
__global__ __launch_bounds__(256) void stats_kernel(const float* __restrict__ x,
                                                    float* __restrict__ mean,
                                                    float* __restrict__ rstd,
                                                    float* __restrict__ pooled) {
  int bg = blockIdx.x;
  int b = bg >> 5, g = bg & 31;
  int ch = threadIdx.x >> 5, lane = threadIdx.x & 31;
  int c = g * 8 + ch;
  const float* p = x + ((size_t)b * CCH + c) * NHW;
  float s = 0.f, s2 = 0.f;
  for (int n = lane; n < NHW; n += 32) {
    float v = p[n];
    s += v; s2 += v * v;
  }
#pragma unroll
  for (int m = 16; m >= 1; m >>= 1) {
    s  += __shfl_xor(s, m, 32);
    s2 += __shfl_xor(s2, m, 32);
  }
  __shared__ float cs[8], cs2[8];
  if (lane == 0) {
    cs[ch] = s; cs2[ch] = s2;
    pooled[b * CCH + c] = s * (1.f / (float)NHW);
  }
  __syncthreads();
  if (threadIdx.x == 0) {
    float S = 0.f, S2 = 0.f;
#pragma unroll
    for (int i = 0; i < 8; ++i) { S += cs[i]; S2 += cs2[i]; }
    float mu = S * (1.f / 8192.f);
    float var = S2 * (1.f / 8192.f) - mu * mu;
    mean[bg] = mu;
    rstd[bg] = rsqrtf(var + 1e-5f);
  }
}

// SE MLP: gate[b][c] = sigmoid(w2 @ relu(w1 @ pooled + b1) + b2). grid = B, block = 256
__global__ __launch_bounds__(256) void se_kernel(const float* __restrict__ pooled,
                                                 const float* __restrict__ w1, const float* __restrict__ b1,
                                                 const float* __restrict__ w2, const float* __restrict__ b2,
                                                 float* __restrict__ gate) {
  int b = blockIdx.x;
  __shared__ float pl[CCH], h1[64];
  pl[threadIdx.x] = pooled[b * CCH + threadIdx.x];
  __syncthreads();
  if (threadIdx.x < 64) {
    float a = b1[threadIdx.x];
    const float* wr = w1 + threadIdx.x * CCH;
    for (int c = 0; c < CCH; ++c) a += wr[c] * pl[c];
    h1[threadIdx.x] = fmaxf(a, 0.f);
  }
  __syncthreads();
  float a = b2[threadIdx.x];
  const float* wr = w2 + threadIdx.x * 64;
  for (int k = 0; k < 64; ++k) a += wr[k] * h1[k];
  gate[b * CCH + threadIdx.x] = 1.f / (1.f + __expf(-a));
}

// GroupNorm apply + transpose: x[b][c][n] -> xnT[b][n][c] in bf16. grid (N/32, C/32, B), block (32,8)
__global__ void norm_transpose_kernel(const float* __restrict__ x,
                                      const float* __restrict__ gamma, const float* __restrict__ beta,
                                      const float* __restrict__ mean, const float* __restrict__ rstd,
                                      __bf16* __restrict__ xnT) {
  __shared__ float tile[32][33];
  int b = blockIdx.z, n0 = blockIdx.x * 32, c0 = blockIdx.y * 32;
  for (int cc = threadIdx.y; cc < 32; cc += 8) {
    int c = c0 + cc, n = n0 + threadIdx.x;
    float v = x[((size_t)b * CCH + c) * NHW + n];
    int g = c >> 3;
    float mu = mean[b * 32 + g], rs = rstd[b * 32 + g];
    tile[cc][threadIdx.x] = (v - mu) * rs * gamma[c] + beta[c];
  }
  __syncthreads();
  for (int nn = threadIdx.y; nn < 32; nn += 8) {
    int n = n0 + nn, c = c0 + threadIdx.x;
    xnT[((size_t)b * NHW + n) * CCH + c] = (__bf16)tile[threadIdx.x][nn];
  }
}

// v-part of qkvT ([n][768], cols 512..767) -> vmat[b][c][n]. grid (N/32, C/32, B), block (32,8)
__global__ void transpose_v_kernel(const __bf16* __restrict__ qkvT, __bf16* __restrict__ vmat) {
  __shared__ __bf16 t[32][33];
  int b = blockIdx.z, n0 = blockIdx.x * 32, c0 = blockIdx.y * 32;
  const __bf16* src = qkvT + (size_t)b * NHW * O3 + 512;
  for (int nn = threadIdx.y; nn < 32; nn += 8)
    t[nn][threadIdx.x] = src[(size_t)(n0 + nn) * O3 + c0 + threadIdx.x];
  __syncthreads();
  __bf16* dst = vmat + (size_t)b * CCH * NHW;
  for (int cc = threadIdx.y; cc < 32; cc += 8)
    dst[(size_t)(c0 + cc) * NHW + n0 + threadIdx.x] = t[threadIdx.x][cc];
}

// ---------------- generic NT bf16 WMMA GEMM ----------------
// D[M x Nn] = A[M x K] (row-major) x BT[Nn x K]^T (NT). 256 threads, block tile 128x64, wave tile 32x32.
// EPI 0: +bias[col], store bf16   EPI 1: store bf16   EPI 2: d_out = resid + (acc+bias[col])*gate (final)
template <int EPI>
__global__ __launch_bounds__(256) void gemm_nt_kernel(
    const __bf16* __restrict__ A, int lda, size_t strideA,
    const __bf16* __restrict__ BT, int ldb, size_t strideB,
    int Kk,
    const float* __restrict__ bias,
    __bf16* __restrict__ obf, int ldo, size_t strideO,
    float* __restrict__ ofl,
    const float* __restrict__ resid,
    const float* __restrict__ gate) {
  int bz = blockIdx.z;
  const __bf16* Ab = A + (size_t)bz * strideA;
  const __bf16* Bb = BT + (size_t)bz * strideB;
  int w = threadIdx.x >> 5, lane = threadIdx.x & 31;
  int m0 = blockIdx.x * 128 + (w & 3) * 32;
  int n0 = blockIdx.y * 64 + (w >> 2) * 32;

  v8f acc[2][2];
  acc[0][0] = vzero8(); acc[0][1] = vzero8();
  acc[1][0] = vzero8(); acc[1][1] = vzero8();

  for (int k0 = 0; k0 < Kk; k0 += 32) {
    v16bf a0 = frag_a(Ab, m0, k0, lda);
    v16bf a1 = frag_a(Ab, m0 + 16, k0, lda);
    v16bf b0 = frag_b(Bb, n0, k0, ldb);
    v16bf b1 = frag_b(Bb, n0 + 16, k0, ldb);
    acc[0][0] = WMMA_BF16(a0, b0, acc[0][0]);
    acc[0][1] = WMMA_BF16(a0, b1, acc[0][1]);
    acc[1][0] = WMMA_BF16(a1, b0, acc[1][0]);
    acc[1][1] = WMMA_BF16(a1, b1, acc[1][1]);
  }

  int cb = lane & 15;
  int rb = (lane < 16) ? 0 : 8;
#pragma unroll
  for (int ti = 0; ti < 2; ++ti)
#pragma unroll
    for (int tj = 0; tj < 2; ++tj) {
      int col = n0 + tj * 16 + cb;
#pragma unroll
      for (int r = 0; r < 8; ++r) {
        int rowi = m0 + ti * 16 + rb + r;
        float v = acc[ti][tj][r];
        if (EPI == 0) {
          v += bias[col];
          obf[(size_t)bz * strideO + (size_t)rowi * ldo + col] = (__bf16)v;
        } else if (EPI == 1) {
          obf[(size_t)bz * strideO + (size_t)rowi * ldo + col] = (__bf16)v;
        } else {
          float g = gate[bz * CCH + col];
          size_t oi = ((size_t)(bz * CCH + col)) * NHW + rowi;
          ofl[oi] = resid[oi] + (v + bias[col]) * g;
        }
      }
    }
}

// ---------------- fused scores + softmax ----------------
// grid (N/16 strips, B), block 256 (8 waves x 128 cols each). S strip 16x1024 f32 in LDS.
__global__ __launch_bounds__(256) void scores_softmax_kernel(const __bf16* __restrict__ qkvT,
                                                             __bf16* __restrict__ P) {
  __shared__ float S[16 * NHW];  // 64 KB
  int b = blockIdx.y;
  int i0 = blockIdx.x * 16;
  const __bf16* Q  = qkvT + (size_t)b * NHW * O3;  // A: rows i, cols c, lda=768
  const __bf16* Kt = Q + 256;                      // BT: rows j, cols c
  int w = threadIdx.x >> 5, lane = threadIdx.x & 31;

  v16bf afr[8];
#pragma unroll
  for (int ks = 0; ks < 8; ++ks) afr[ks] = frag_a(Q, i0, ks * 32, O3);

#pragma unroll
  for (int jt = 0; jt < 8; ++jt) {
    int j0 = w * 128 + jt * 16;
    v8f acc = vzero8();
#pragma unroll
    for (int ks = 0; ks < 8; ++ks) {
      v16bf bfr = frag_b(Kt, j0, ks * 32, O3);
      acc = WMMA_BF16(afr[ks], bfr, acc);
    }
    int col = j0 + (lane & 15);
    int rbse = (lane < 16) ? 0 : 8;
#pragma unroll
    for (int r = 0; r < 8; ++r) S[(rbse + r) * NHW + col] = acc[r] * 0.0625f;  // 1/sqrt(256)
  }
  __syncthreads();

  // softmax: 16 rows x 16 threads/row, each thread strides 16 over 1024 cols
  int rrow = threadIdx.x >> 4;
  int t = threadIdx.x & 15;
  float* row = S + rrow * NHW;
  float mx = -3.402823466e38f;
  for (int k = 0; k < 64; ++k) mx = fmaxf(mx, row[t + k * 16]);
#pragma unroll
  for (int m = 8; m >= 1; m >>= 1) mx = fmaxf(mx, __shfl_xor(mx, m, 32));
  float sum = 0.f;
  for (int k = 0; k < 64; ++k) {
    int c = t + k * 16;
    float e = __expf(row[c] - mx);
    row[c] = e;
    sum += e;
  }
#pragma unroll
  for (int m = 8; m >= 1; m >>= 1) sum += __shfl_xor(sum, m, 32);
  float inv = 1.f / sum;
  __bf16* Pr = P + (size_t)b * NHW * NHW + (size_t)(i0 + rrow) * NHW;
  for (int k = 0; k < 64; ++k) {
    int c = t + k * 16;
    Pr[c] = (__bf16)(row[c] * inv);
  }
}

// ---------------- host side ----------------
extern "C" void kernel_launch(void* const* d_in, const int* in_sizes, int n_in,
                              void* d_out, int out_size, void* d_ws, size_t ws_size,
                              hipStream_t stream) {
  const float* x       = (const float*)d_in[0];
  const float* gamma   = (const float*)d_in[1];
  const float* beta    = (const float*)d_in[2];
  const float* w_qkv   = (const float*)d_in[3];
  const float* b_qkv   = (const float*)d_in[4];
  const float* w_proj  = (const float*)d_in[5];
  const float* b_proj  = (const float*)d_in[6];
  const float* w_se1   = (const float*)d_in[7];
  const float* b_se1   = (const float*)d_in[8];
  const float* w_se2   = (const float*)d_in[9];
  const float* b_se2   = (const float*)d_in[10];
  float* out = (float*)d_out;

  char* ws = (char*)d_ws;
  size_t off = 0;
  auto take = [&](size_t bytes) { char* p = ws + off; off += (bytes + 255) & ~(size_t)255; return p; };

  __bf16* wqkv_bf  = (__bf16*)take((size_t)O3 * CCH * 2);        // 384 KB
  __bf16* wproj_bf = (__bf16*)take((size_t)CCH * CCH * 2);       // 128 KB
  float*  meanw    = (float*)take((size_t)BSZ * 32 * 4);
  float*  rstdw    = (float*)take((size_t)BSZ * 32 * 4);
  float*  pooled   = (float*)take((size_t)BSZ * CCH * 4);
  float*  gatew    = (float*)take((size_t)BSZ * CCH * 4);
  __bf16* xnT_OT   = (__bf16*)take((size_t)BSZ * NHW * CCH * 2); // 16 MB (xnT, reused as O_T)
  __bf16* qkvT     = (__bf16*)take((size_t)BSZ * NHW * O3 * 2);  // 48 MB
  __bf16* vmat     = (__bf16*)take((size_t)BSZ * CCH * NHW * 2); // 16 MB
  __bf16* P        = (__bf16*)take((size_t)BSZ * NHW * NHW * 2); // 64 MB

  // 1. weight conversion to bf16 (original [o][c] layouts serve as NT "BT" operands)
  cvt_bf16_kernel<<<(O3 * CCH + 255) / 256, 256, 0, stream>>>(w_qkv, wqkv_bf, O3 * CCH);
  cvt_bf16_kernel<<<(CCH * CCH + 255) / 256, 256, 0, stream>>>(w_proj, wproj_bf, CCH * CCH);

  // 2. GroupNorm stats + SE pooled means
  stats_kernel<<<BSZ * 32, 256, 0, stream>>>(x, meanw, rstdw, pooled);

  // 3. SE gate
  se_kernel<<<BSZ, 256, 0, stream>>>(pooled, w_se1, b_se1, w_se2, b_se2, gatew);

  // 4. normalize + transpose -> xnT[b][n][c] bf16
  norm_transpose_kernel<<<dim3(NHW / 32, CCH / 32, BSZ), dim3(32, 8), 0, stream>>>(
      x, gamma, beta, meanw, rstdw, xnT_OT);

  // 5. QKV GEMM: qkvT[b][n][o] = xnT x w_qkv^T + b_qkv
  gemm_nt_kernel<0><<<dim3(NHW / 128, O3 / 64, BSZ), 256, 0, stream>>>(
      xnT_OT, CCH, (size_t)NHW * CCH, wqkv_bf, CCH, 0, CCH,
      b_qkv, qkvT, O3, (size_t)NHW * O3, nullptr, nullptr, nullptr);

  // 6. transpose V -> vmat[b][c][n]
  transpose_v_kernel<<<dim3(NHW / 32, CCH / 32, BSZ), dim3(32, 8), 0, stream>>>(qkvT, vmat);

  // 7. scores (Q^T K * scale) + softmax -> P[b][i][j] bf16
  scores_softmax_kernel<<<dim3(NHW / 16, BSZ), 256, 0, stream>>>(qkvT, P);

  // 8. O_T[b][i][c] = P x V_T  (BT = vmat)
  gemm_nt_kernel<1><<<dim3(NHW / 128, CCH / 64, BSZ), 256, 0, stream>>>(
      P, NHW, (size_t)NHW * NHW, vmat, NHW, (size_t)CCH * NHW, NHW,
      nullptr, xnT_OT, CCH, (size_t)NHW * CCH, nullptr, nullptr, nullptr);

  // 9. proj GEMM + fused epilogue: out = x + (O_T x w_proj^T + b_proj) * gate
  gemm_nt_kernel<2><<<dim3(NHW / 128, CCH / 64, BSZ), 256, 0, stream>>>(
      xnT_OT, CCH, (size_t)NHW * CCH, wproj_bf, CCH, 0, CCH,
      b_proj, nullptr, 0, 0, out, x, gatew);

  (void)in_sizes; (void)n_in; (void)out_size; (void)ws_size;
}
